// NonLinearNet_62766652064010
// MI455X (gfx1250) — matrix operand order repaired and checked by reading, not compiled, not run on hardware
//
#include <hip/hip_runtime.h>
#include <hip/hip_bf16.h>

typedef __attribute__((ext_vector_type(16))) __bf16 bf16x16;
typedef __attribute__((ext_vector_type(8)))  __bf16 bf16x8;
typedef __attribute__((ext_vector_type(8)))  float  f32x8;
typedef __attribute__((ext_vector_type(8)))  float  v8f;

#define MLP_BATCH 524288
#define WAVES_PER_BLOCK 8

__device__ __forceinline__ bf16x16 make_afrag(bf16x8 lo, bf16x8 hi) {
  return __builtin_shufflevector(lo, hi, 0, 1, 2, 3, 4, 5, 6, 7,
                                 8, 9, 10, 11, 12, 13, 14, 15);
}

// Load one 16x32 bf16 A fragment from an LDS activation buffer with row
// stride KP (contiguous-k A order used here, matched by B: lane holds row lane&15; elems 0..15 =
// K koff+{0..15}, koff = (lane>>4)*16).
template <int KP>
__device__ __forceinline__ bf16x16 load_afrag(const __bf16* __restrict__ in,
                                              int kbase, int lane) {
  const int row = lane & 15;
  const int koff = (lane >> 4) * 16;
  const __bf16* p = in + row * KP + kbase;
  bf16x8 lo = *(const bf16x8*)(p + koff);
  bf16x8 hi = *(const bf16x8*)(p + koff + 8);
  return make_afrag(lo, hi);
}

// Bias + ReLU + bf16 store of one f32 C tile into an LDS act buffer with row
// stride SOUT (C layout: lane = column n, VGPR r = row r + 8*(lane>>4)).
template <int SOUT, bool RELU>
__device__ __forceinline__ void store_ctile(__bf16* __restrict__ out, v8f acc,
                                            float bv, int n, int lane) {
  const int mbase = (lane >> 4) * 8;
#pragma unroll
  for (int r = 0; r < 8; ++r) {
    float v = acc[r] + bv;
    if (RELU) v = v > 0.0f ? v : 0.0f;
    out[(mbase + r) * SOUT + n] = (__bf16)v;
  }
}

// Hidden layer with loop-invariant B fragments held in registers (tail).
template <int KS, int NT>
__device__ __forceinline__ void layer_regB(const __bf16* __restrict__ in,
                                           __bf16* __restrict__ out,
                                           const bf16x16 (&B)[NT][KS],
                                           const float (&bv)[NT], int lane) {
  constexpr int KP = KS * 32;
  constexpr int SOUT = NT * 16;
  bf16x16 a[KS];
#pragma unroll
  for (int ks = 0; ks < KS; ++ks) a[ks] = load_afrag<KP>(in, ks * 32, lane);
#pragma unroll
  for (int nt = 0; nt < NT; ++nt) {
    v8f acc = {};
#pragma unroll
    for (int ks = 0; ks < KS; ++ks)
      acc = __builtin_amdgcn_wmma_f32_16x16x32_bf16(
          false, a[ks], false, B[nt][ks], (short)0, acc, false, false);
    store_ctile<SOUT, true>(out, acc, bv[nt], nt * 16 + (lane & 15), lane);
  }
}

// Hidden layer with B fragments streamed from LDS weights (big layers).
template <int KS, int NT>
__device__ __forceinline__ void layer_ldsB(const __bf16* __restrict__ in,
                                           __bf16* __restrict__ out,
                                           const __bf16* __restrict__ w,
                                           const float (&bv)[NT], int lane) {
  constexpr int KP = KS * 32;
  constexpr int SOUT = NT * 16;
  const int row = lane & 15;
  const int half = lane >> 4;
  bf16x16 a[KS];
#pragma unroll
  for (int ks = 0; ks < KS; ++ks) a[ks] = load_afrag<KP>(in, ks * 32, lane);
#pragma unroll
  for (int nt = 0; nt < NT; ++nt) {
    v8f acc = {};
#pragma unroll
    for (int ks = 0; ks < KS; ++ks) {
      bf16x16 b =
          *(const bf16x16*)(w + (nt * 16 + row) * KP + ks * 32 + half * 16);
      acc = __builtin_amdgcn_wmma_f32_16x16x32_bf16(
          false, a[ks], false, b, (short)0, acc, false, false);
    }
    store_ctile<SOUT, true>(out, acc, bv[nt], nt * 16 + row, lane);
  }
}

__global__ void __launch_bounds__(256, 4) mlp7_wmma_kernel(
    const float* __restrict__ x,
    const float* __restrict__ W1, const float* __restrict__ b1,
    const float* __restrict__ W2, const float* __restrict__ b2,
    const float* __restrict__ W3, const float* __restrict__ b3,
    const float* __restrict__ W4, const float* __restrict__ b4,
    const float* __restrict__ W5, const float* __restrict__ b5,
    const float* __restrict__ W6, const float* __restrict__ b6,
    const float* __restrict__ W7, const float* __restrict__ b7,
    float* __restrict__ out) {
  // Padded layer geometry. KP[l] = K padded to mult of 32; NP[l] = N padded so
  // that NP[l] == KP[l+1] (next layer's K-padding is always freshly zeroed:
  // padded weight rows/bias are 0 and ReLU(0) = 0).
  // l:      1    2    3    4    5    6    7
  // dIn : 128   84   42   32   16    8    4
  // dOut:  84   42   32   16    8    4    1
  // KP  : 128   96   64   32   32   32   32
  // NP  :  96   64   32   32   32   32   16
  __shared__ __attribute__((aligned(32))) __bf16 sW[24064];
  __shared__ float sB[304];
  __shared__ __attribute__((aligned(32))) __bf16 sActA[WAVES_PER_BLOCK][16 * 96];
  __shared__ __attribute__((aligned(32))) __bf16 sActB[WAVES_PER_BLOCK][16 * 64];

  {
    const float* Wp[7] = {W1, W2, W3, W4, W5, W6, W7};
    const float* Bp[7] = {b1, b2, b3, b4, b5, b6, b7};
    const int dIn[7]  = {128, 84, 42, 32, 16, 8, 4};
    const int dOut[7] = {84, 42, 32, 16, 8, 4, 1};
    const int KPs[7]  = {128, 96, 64, 32, 32, 32, 32};
    const int NPs[7]  = {96, 64, 32, 32, 32, 32, 16};
    const int wOff[7] = {0, 12288, 18432, 20480, 21504, 22528, 23552};
    const int bOff[7] = {0, 96, 160, 192, 224, 256, 288};
#pragma unroll
    for (int l = 0; l < 7; ++l) {
      const int kp = KPs[l];
      const int tot = NPs[l] * kp;
      for (int i = threadIdx.x; i < tot; i += 256) {
        const int n = i / kp;
        const int k = i - n * kp;
        float v = (n < dOut[l] && k < dIn[l]) ? Wp[l][n * dIn[l] + k] : 0.0f;
        sW[wOff[l] + i] = (__bf16)v;
      }
      for (int i = threadIdx.x; i < NPs[l]; i += 256)
        sB[bOff[l] + i] = (i < dOut[l]) ? Bp[l][i] : 0.0f;
    }
  }
  __syncthreads();

  const int wave = threadIdx.x >> 5;
  const int lane = threadIdx.x & 31;
  const int row16 = lane & 15;
  const int half = lane >> 4;
  const int koff = half * 16;
  __bf16* actA = sActA[wave];
  __bf16* actB = sActB[wave];

  // ---- Hoist loop-invariant operands into registers -----------------------
  // Biases (per-lane scalar for column n = nt*16 + row16).
  float bv1[6], bv2[4], bv3[2], bv4[2], bv5[2], bv6[2];
#pragma unroll
  for (int nt = 0; nt < 6; ++nt) bv1[nt] = sB[0 + nt * 16 + row16];
#pragma unroll
  for (int nt = 0; nt < 4; ++nt) bv2[nt] = sB[96 + nt * 16 + row16];
#pragma unroll
  for (int nt = 0; nt < 2; ++nt) {
    bv3[nt] = sB[160 + nt * 16 + row16];
    bv4[nt] = sB[192 + nt * 16 + row16];
    bv5[nt] = sB[224 + nt * 16 + row16];
    bv6[nt] = sB[256 + nt * 16 + row16];
  }
  const float bv7 = sB[288];  // only column 0 is consumed

  // B fragments for the serial tail layers 4..7 only (7 frags = 56 VGPRs);
  // layers 1-3 stream B from LDS to keep allocation under 256 VGPRs.
  bf16x16 B4[2][1], B5[2][1], B6[2][1], B7;
#pragma unroll
  for (int nt = 0; nt < 2; ++nt) {
    B4[nt][0] = *(const bf16x16*)(sW + 20480 + (nt * 16 + row16) * 32 + half * 16);
    B5[nt][0] = *(const bf16x16*)(sW + 21504 + (nt * 16 + row16) * 32 + half * 16);
    B6[nt][0] = *(const bf16x16*)(sW + 22528 + (nt * 16 + row16) * 32 + half * 16);
  }
  B7 = *(const bf16x16*)(sW + 23552 + row16 * 32 + half * 16);

  const int nTilesTotal = MLP_BATCH / 16;  // 32768 row-tiles of 16
  const int tileStride = gridDim.x * WAVES_PER_BLOCK;

  for (int tile = blockIdx.x * WAVES_PER_BLOCK + wave; tile < nTilesTotal;
       tile += tileStride) {
    const int rowbase = tile * 16;

    // Prefetch next tile's x block (8 KB, 256 B per lane).
    {
      const int nextTile = tile + tileStride;
      if (nextTile < nTilesTotal) {
        const char* np = (const char*)(x + (size_t)nextTile * 16 * 128);
        __builtin_prefetch(np + lane * 256, 0, 0);
      }
    }

    // ---------------- Layer 1: A from global fp32, 4 K-steps, 6 N-tiles ----
    bf16x16 a1[4];
    {
      const float* xr = x + (size_t)(rowbase + row16) * 128;
#pragma unroll
      for (int ks = 0; ks < 4; ++ks) {
        f32x8 lo = *(const f32x8*)(xr + ks * 32 + koff);
        f32x8 hi = *(const f32x8*)(xr + ks * 32 + koff + 8);
        bf16x16 a;
#pragma unroll
        for (int e = 0; e < 8; ++e) {
          a[e] = (__bf16)lo[e];
          a[e + 8] = (__bf16)hi[e];
        }
        a1[ks] = a;
      }
    }
#pragma unroll
    for (int nt = 0; nt < 6; ++nt) {
      v8f acc = {};
#pragma unroll
      for (int ks = 0; ks < 4; ++ks) {
        bf16x16 b =
            *(const bf16x16*)(sW + (nt * 16 + row16) * 128 + ks * 32 + half * 16);
        acc = __builtin_amdgcn_wmma_f32_16x16x32_bf16(
            false, a1[ks], false, b, (short)0, acc, false, false);
      }
      store_ctile<96, true>(actA, acc, bv1[nt], nt * 16 + row16, lane);
    }

    // ---------------- Layers 2..6 (LDS ping-pong) ---------------------------
    layer_ldsB<3, 4>(actA, actB, sW + 12288, bv2, lane);  // 84->42
    layer_ldsB<2, 2>(actB, actA, sW + 18432, bv3, lane);  // 42->32
    layer_regB<1, 2>(actA, actB, B4, bv4, lane);          // 32->16
    layer_regB<1, 2>(actB, actA, B5, bv5, lane);          // 16->8
    layer_regB<1, 2>(actA, actB, B6, bv6, lane);          // 8->4

    // ---------------- Layer 7: 4->1, single WMMA, write column 0 ------------
    {
      bf16x16 a = load_afrag<32>(actB, 0, lane);
      v8f acc = {};
      acc = __builtin_amdgcn_wmma_f32_16x16x32_bf16(
          false, a, false, B7, (short)0, acc, false, false);
      if (row16 == 0) {
        const int mbase = half * 8;
#pragma unroll
        for (int r = 0; r < 8; ++r) out[rowbase + mbase + r] = acc[r] + bv7;
      }
    }
  }
}

extern "C" void kernel_launch(void* const* d_in, const int* in_sizes, int n_in,
                              void* d_out, int out_size, void* d_ws,
                              size_t ws_size, hipStream_t stream) {
  (void)in_sizes; (void)n_in; (void)out_size; (void)d_ws; (void)ws_size;
  const float* x  = (const float*)d_in[0];
  const float* W1 = (const float*)d_in[1];
  const float* b1 = (const float*)d_in[2];
  const float* W2 = (const float*)d_in[3];
  const float* b2 = (const float*)d_in[4];
  const float* W3 = (const float*)d_in[5];
  const float* b3 = (const float*)d_in[6];
  const float* W4 = (const float*)d_in[7];
  const float* b4 = (const float*)d_in[8];
  const float* W5 = (const float*)d_in[9];
  const float* b5 = (const float*)d_in[10];
  const float* W6 = (const float*)d_in[11];
  const float* b6 = (const float*)d_in[12];
  const float* W7 = (const float*)d_in[13];
  const float* b7 = (const float*)d_in[14];
  float* out = (float*)d_out;

  // 1024 blocks x 8 waves x 16 rows = 131072 rows/pass; 4 grid-stride passes.
  mlp7_wmma_kernel<<<dim3(1024), dim3(256), 0, stream>>>(
      x, W1, b1, W2, b2, W3, b3, W4, b4, W5, b5, W6, b6, W7, b7, out);
}
